// CrossAttention_85418309582869
// MI455X (gfx1250) — matrix operand order repaired and checked by reading, not compiled, not run on hardware
//
#include <hip/hip_runtime.h>
#include <hip/hip_bf16.h>

typedef _Float16 h16;
typedef __attribute__((ext_vector_type(16))) _Float16 v16h;
typedef __attribute__((ext_vector_type(4)))  _Float16 v4h;
typedef __attribute__((ext_vector_type(8)))  float    v8f;

__device__ __forceinline__ v8f zero8() {
    v8f z = {0.f,0.f,0.f,0.f,0.f,0.f,0.f,0.f};
    return z;
}

// A-fragment (16 x 32 chunk, f16): lane (0-15)=row m, half selects K 0..7/16..23 vs 8..15/24..31
// base points at row-major [16][stride] f16, K-chunk start.
__device__ __forceinline__ v16h load_A(const h16* base, int stride, int lane) {
    union { v16h v; unsigned u[8]; } a;
    int m = lane & 15, half = lane >> 4;
#pragma unroll
    for (int i = 0; i < 8; ++i) {
        int k = ((i < 4) ? (2 * i) : (16 + 2 * (i - 4))) + 8 * half;
        __builtin_memcpy(&a.u[i], base + m * stride + k, 4);
    }
    return a.v;
}

// B-fragment (32x16, f16) from transposed layout BT[n][k] (so B[k][n] = BT[n][k]):
// half selects K 0..7/16..23 vs 8..15/24..31 (same order as load_A).
__device__ __forceinline__ v16h load_BT(const h16* base, int stride, int lane) {
    union { v16h v; unsigned u[8]; } b;
    int n = lane & 15, khalf = lane >> 4;
#pragma unroll
    for (int i = 0; i < 8; ++i) {
        int k = ((i < 4) ? (2 * i) : (16 + 2 * (i - 4))) + 8 * khalf;
        __builtin_memcpy(&b.u[i], base + n * stride + k, 4);
    }
    return b.v;
}

__device__ __forceinline__ v8f wmma_f16(v16h a, v16h b, v8f c) {
    return __builtin_amdgcn_wmma_f32_16x16x32_f16(false, a, false, b, (short)0, c, false, false);
}

// ---------- prep kernels ----------
__global__ void f32_to_f16_kernel(const float4* __restrict__ src, h16* __restrict__ dst, int n4) {
    int i = blockIdx.x * blockDim.x + threadIdx.x;
    int stride = gridDim.x * blockDim.x;
    for (; i < n4; i += stride) {
        float4 v = src[i];
        v4h h = { (h16)v.x, (h16)v.y, (h16)v.z, (h16)v.w };
        *(v4h*)(dst + 4 * i) = h;
    }
}

// dst[dR][dC] f16 with dst[n][k] = src[k][n] (src is [R][C] f32), zero padded.
__global__ void transpose_to_f16_kernel(const float* __restrict__ src, h16* __restrict__ dst,
                                        int R, int C, int dR, int dC) {
    int idx = blockIdx.x * blockDim.x + threadIdx.x;
    if (idx >= dR * dC) return;
    int n = idx / dC, k = idx % dC;
    dst[idx] = (h16)((n < C && k < R) ? src[k * C + n] : 0.f);
}

// ---------- K/V projection: enc[b,77,768] @ Wk/Wv -> Kpad[b,h,80,64], Vt[b,h,48,96] ----------
__global__ void kv_proj_kernel(const float* __restrict__ enc,
                               const h16* __restrict__ WkT,   // [320][768]
                               const h16* __restrict__ WvT,   // [320][768]
                               h16* __restrict__ Kpad,        // [B,H,80,64]
                               h16* __restrict__ Vt)          // [B,H,48,96]
{
    __shared__ __align__(16) h16 encT[16 * 776];
    int b = blockIdx.x / 5, mt = blockIdx.x % 5;
    int tid = threadIdx.x;
    // stage encoder tile (f32 -> f16), rows >= 77 zeroed; 192 float4 per row
    for (int i = tid; i < 16 * 192; i += 256) {
        int row = i / 192, j = i % 192;
        int tok = mt * 16 + row;
        float4 v = make_float4(0.f, 0.f, 0.f, 0.f);
        if (tok < 77) v = ((const float4*)(enc + ((size_t)b * 77 + tok) * 768))[j];
        v4h h = { (h16)v.x, (h16)v.y, (h16)v.z, (h16)v.w };
        *(v4h*)(encT + row * 776 + j * 4) = h;
    }
    __syncthreads();
    int wave = tid >> 5, lane = tid & 31;
    int n = lane & 15, half = lane >> 4;
    for (int j = wave; j < 40; j += 8) {
        const h16* WT = (j < 20) ? WkT : WvT;
        int nt = j % 20;
        const h16* Wbase = WT + (size_t)(nt * 16) * 768;
        v8f acc = zero8();
        // software-pipelined K loop (24 steps)
        v16h a  = load_A(encT, 776, lane);
        v16h bb = load_BT(Wbase, 768, lane);
        for (int ks = 0; ks < 23; ++ks) {
            v16h a2 = load_A(encT + (ks + 1) * 32, 776, lane);
            v16h b2 = load_BT(Wbase + (ks + 1) * 32, 768, lane);
            acc = wmma_f16(a, bb, acc);
            a = a2; bb = b2;
        }
        acc = wmma_f16(a, bb, acc);
#pragma unroll
        for (int r = 0; r < 8; ++r) {
            int m = r + 8 * half;
            int tok = mt * 16 + m;
            int c = nt * 16 + n;
            int h = c / 40, d = c % 40;
            h16 val = (h16)acc[r];
            if (j < 20) Kpad[(((size_t)(b * 8 + h) * 80) + tok) * 64 + d] = val;
            else        Vt[(((size_t)(b * 8 + h) * 48) + d) * 96 + tok]   = val;
        }
    }
}

// ---------- fused Q-proj + softmax(QK^T) + PV ----------
__global__ void attn_kernel(const h16* __restrict__ hs,    // [B,4096,320] f16
                            const h16* __restrict__ WqTp,  // [336][320] f16 (zero padded rows)
                            const h16* __restrict__ Kpad,  // [B,H,80,64]
                            const h16* __restrict__ Vt,    // [B,H,48,96]
                            h16* __restrict__ attn)        // [B,4096,320]
{
    __shared__ __align__(16) h16 Kl[80 * 64];
    __shared__ __align__(16) h16 Vl[48 * 96];
    __shared__ __align__(16) h16 Ql[8][16 * 64];
    __shared__ __align__(16) h16 Pl[8][16 * 96];

    int b = blockIdx.x >> 8, h = (blockIdx.x >> 5) & 7, chunk = blockIdx.x & 31;
    int tid = threadIdx.x, wave = tid >> 5, lane = tid & 31;
    int n = lane & 15, half = lane >> 4;

    const uint4* Kg4 = (const uint4*)(Kpad + (size_t)(b * 8 + h) * 80 * 64);
    const uint4* Vg4 = (const uint4*)(Vt   + (size_t)(b * 8 + h) * 48 * 96);
    for (int i = tid; i < 640; i += 256) ((uint4*)Kl)[i] = Kg4[i];  // 80*64*2B = 640*16B
    for (int i = tid; i < 576; i += 256) ((uint4*)Vl)[i] = Vg4[i];  // 48*96*2B = 576*16B
    __syncthreads();

    int qtile = chunk * 8 + wave;
    int r0 = qtile * 16;
    const h16* hsb = hs + ((size_t)b * 4096 + r0) * 320;
    const h16* W0 = WqTp + (size_t)(h * 40 + 0)  * 320;
    const h16* W1 = WqTp + (size_t)(h * 40 + 16) * 320;
    const h16* W2 = WqTp + (size_t)(h * 40 + 32) * 320;

    // ---- Q projection: q[16][48] = hs_tile[16][320] @ Wq[:, h*40 .. h*40+48)  (pipelined)
    v8f accq0 = zero8(), accq1 = zero8(), accq2 = zero8();
    v16h a  = load_A(hsb, 320, lane);
    v16h b0 = load_BT(W0, 320, lane);
    v16h b1 = load_BT(W1, 320, lane);
    v16h b2 = load_BT(W2, 320, lane);
    for (int ks = 0; ks < 9; ++ks) {
        int o = (ks + 1) * 32;
        v16h an = load_A(hsb + o, 320, lane);
        v16h c0 = load_BT(W0 + o, 320, lane);
        v16h c1 = load_BT(W1 + o, 320, lane);
        v16h c2 = load_BT(W2 + o, 320, lane);
        accq0 = wmma_f16(a, b0, accq0);
        accq1 = wmma_f16(a, b1, accq1);
        accq2 = wmma_f16(a, b2, accq2);
        a = an; b0 = c0; b1 = c1; b2 = c2;
    }
    accq0 = wmma_f16(a, b0, accq0);
    accq1 = wmma_f16(a, b1, accq1);
    accq2 = wmma_f16(a, b2, accq2);

    // transpose q through LDS into A-layout [16][64], zero cols >= 40
#pragma unroll
    for (int r = 0; r < 8; ++r) {
        int m = r + 8 * half;
        Ql[wave][m * 64 + 0  + n] = (h16)accq0[r];
        Ql[wave][m * 64 + 16 + n] = (h16)accq1[r];
        Ql[wave][m * 64 + 32 + n] = ((32 + n) < 40) ? (h16)accq2[r] : (h16)0.f;
        Ql[wave][m * 64 + 48 + n] = (h16)0.f;
    }
    asm volatile("s_wait_dscnt 0" ::: "memory");

    // ---- scores[16][80] = q @ K^T  (B[k][n] = K[token n][dim k] = Kl[n][k])
    v16h aq0 = load_A(Ql[wave], 64, lane);
    v16h aq1 = load_A(Ql[wave] + 32, 64, lane);
    v8f s[5];
#pragma unroll
    for (int nt = 0; nt < 5; ++nt) {
        s[nt] = zero8();
        v16h k0 = load_BT(Kl + nt * 16 * 64, 64, lane);
        s[nt] = wmma_f16(aq0, k0, s[nt]);
        v16h k1 = load_BT(Kl + nt * 16 * 64 + 32, 64, lane);
        s[nt] = wmma_f16(aq1, k1, s[nt]);
    }

    // ---- softmax over 77 tokens (row = r + 8*half, striped over lanes 0-15 per half)
    const float scale = 0.15811388300841898f; // 40^-0.5
#pragma unroll
    for (int r = 0; r < 8; ++r) {
        int m = r + 8 * half;
        float vv[5];
        float mx = -1e30f;
#pragma unroll
        for (int nt = 0; nt < 5; ++nt) {
            int tok = nt * 16 + n;
            float x = (tok < 77) ? (float)s[nt][r] * scale : -1e30f;
            vv[nt] = x;
            mx = fmaxf(mx, x);
        }
        for (int msk = 1; msk < 16; msk <<= 1) mx = fmaxf(mx, __shfl_xor(mx, msk, 32));
        float sum = 0.f;
#pragma unroll
        for (int nt = 0; nt < 5; ++nt) { vv[nt] = __expf(vv[nt] - mx); sum += vv[nt]; }
        for (int msk = 1; msk < 16; msk <<= 1) sum += __shfl_xor(sum, msk, 32);
        float inv = 1.f / sum;
#pragma unroll
        for (int nt = 0; nt < 5; ++nt)
            Pl[wave][m * 96 + nt * 16 + n] = (h16)(vv[nt] * inv);
        Pl[wave][m * 96 + 80 + n] = (h16)0.f;
    }
    asm volatile("s_wait_dscnt 0" ::: "memory");

    // ---- out[16][48] = P[16][96] @ V[96][48]  (B[k][n] = V[tok k][dim n] = Vl[n][k])
    v8f o0 = zero8(), o1 = zero8(), o2 = zero8();
#pragma unroll
    for (int ks = 0; ks < 3; ++ks) {
        v16h ap = load_A(Pl[wave] + ks * 32, 96, lane);
        v16h bv0 = load_BT(Vl + 0  * 96 + ks * 32, 96, lane);
        v16h bv1 = load_BT(Vl + 16 * 96 + ks * 32, 96, lane);
        v16h bv2 = load_BT(Vl + 32 * 96 + ks * 32, 96, lane);
        o0 = wmma_f16(ap, bv0, o0);
        o1 = wmma_f16(ap, bv1, o1);
        o2 = wmma_f16(ap, bv2, o2);
    }
#pragma unroll
    for (int r = 0; r < 8; ++r) {
        int m = r + 8 * half;
        size_t rowbase = ((size_t)b * 4096 + r0 + m) * 320 + h * 40;
        attn[rowbase + 0  + n] = (h16)o0[r];
        attn[rowbase + 16 + n] = (h16)o1[r];
        if ((32 + n) < 40) attn[rowbase + 32 + n] = (h16)o2[r];
    }
}

// ---------- out projection: attn[b,4096,320] @ Wo + bo -> f32 out ----------
__global__ void out_proj_kernel(const h16* __restrict__ attn,
                                const h16* __restrict__ WoT,   // [320][320]
                                const float* __restrict__ bo,
                                float* __restrict__ out)
{
    __shared__ __align__(16) h16 Al[16 * 328];
    int b = blockIdx.x >> 8, mt = blockIdx.x & 255;
    int tid = threadIdx.x;
    const h16* src = attn + ((size_t)b * 4096 + mt * 16) * 320;
    // 16-byte vector staging: 40 chunks of 8 f16 per row
    for (int i = tid; i < 16 * 40; i += 256) {
        int row = i / 40, j = i % 40;
        *(uint4*)(Al + row * 328 + j * 8) = *(const uint4*)(src + row * 320 + j * 8);
    }
    __syncthreads();
    int wave = tid >> 5, lane = tid & 31, n = lane & 15, half = lane >> 4;
    for (int nt = wave; nt < 20; nt += 8) {
        const h16* Wbase = WoT + (size_t)(nt * 16) * 320;
        v8f acc = zero8();
        v16h a  = load_A(Al, 328, lane);
        v16h bb = load_BT(Wbase, 320, lane);
        for (int ks = 0; ks < 9; ++ks) {
            int o = (ks + 1) * 32;
            v16h a2 = load_A(Al + o, 328, lane);
            v16h b2 = load_BT(Wbase + o, 320, lane);
            acc = wmma_f16(a, bb, acc);
            a = a2; bb = b2;
        }
        acc = wmma_f16(a, bb, acc);
        float bias = bo[nt * 16 + n];
#pragma unroll
        for (int r = 0; r < 8; ++r) {
            int m = r + 8 * half;
            out[((size_t)b * 4096 + mt * 16 + m) * 320 + nt * 16 + n] = acc[r] + bias;
        }
    }
}

extern "C" void kernel_launch(void* const* d_in, const int* in_sizes, int n_in,
                              void* d_out, int out_size, void* d_ws, size_t ws_size,
                              hipStream_t stream) {
    (void)in_sizes; (void)n_in; (void)out_size; (void)ws_size;
    const float* hs  = (const float*)d_in[0];
    const float* enc = (const float*)d_in[1];
    const float* Wq  = (const float*)d_in[2];
    const float* Wk  = (const float*)d_in[3];
    const float* Wv  = (const float*)d_in[4];
    const float* Wo  = (const float*)d_in[5];
    const float* bo  = (const float*)d_in[6];
    float* out = (float*)d_out;

    char* ws = (char*)d_ws;
    size_t off = 0;
    auto alloc = [&](size_t bytes) -> void* {
        off = (off + 255) & ~(size_t)255;
        void* p = ws + off;
        off += bytes;
        return p;
    };
    h16* hsf  = (h16*)alloc((size_t)16 * 4096 * 320 * 2);
    h16* attn = (h16*)alloc((size_t)16 * 4096 * 320 * 2);
    h16* WqTp = (h16*)alloc((size_t)336 * 320 * 2);
    h16* WkT  = (h16*)alloc((size_t)320 * 768 * 2);
    h16* WvT  = (h16*)alloc((size_t)320 * 768 * 2);
    h16* WoT  = (h16*)alloc((size_t)320 * 320 * 2);
    h16* Kpad = (h16*)alloc((size_t)16 * 8 * 80 * 64 * 2);
    h16* Vt   = (h16*)alloc((size_t)16 * 8 * 48 * 96 * 2);

    f32_to_f16_kernel<<<4096, 256, 0, stream>>>((const float4*)hs, hsf, (16 * 4096 * 320) / 4);
    transpose_to_f16_kernel<<<(336 * 320 + 255) / 256, 256, 0, stream>>>(Wq, WqTp, 320, 320, 336, 320);
    transpose_to_f16_kernel<<<(320 * 768 + 255) / 256, 256, 0, stream>>>(Wk, WkT, 768, 320, 320, 768);
    transpose_to_f16_kernel<<<(320 * 768 + 255) / 256, 256, 0, stream>>>(Wv, WvT, 768, 320, 320, 768);
    transpose_to_f16_kernel<<<(320 * 320 + 255) / 256, 256, 0, stream>>>(Wo, WoT, 320, 320, 320, 320);
    hipMemsetAsync(Kpad, 0, (size_t)16 * 8 * 80 * 64 * 2, stream);
    hipMemsetAsync(Vt,   0, (size_t)16 * 8 * 48 * 96 * 2, stream);

    kv_proj_kernel<<<16 * 5, 256, 0, stream>>>(enc, WkT, WvT, Kpad, Vt);
    attn_kernel<<<16 * 8 * 32, 256, 0, stream>>>(hsf, WqTp, Kpad, Vt, attn);
    out_proj_kernel<<<16 * 256, 256, 0, stream>>>(attn, WoT, bo, out);
}
